// MHCrossAttention_46505905881219
// MI455X (gfx1250) — matrix operand order repaired and checked by reading, not compiled, not run on hardware
//
#include <hip/hip_runtime.h>
#include <hip/hip_bf16.h>

typedef __attribute__((ext_vector_type(16))) __bf16 v16bf;
typedef __attribute__((ext_vector_type(8)))  float  v8f;

#define E_DIM 1024
#define HEADS 16
#define HDIM  64
#define T_DIM 2048
#define TC_DIM 2048
#define BATCH 2

// -------------------------------------------------------------------------
// CDNA5 helpers: async global->LDS copy (ASYNCcnt-tracked) + LDS address
// -------------------------------------------------------------------------
__device__ __forceinline__ unsigned lds_addr(const void* p) {
  // generic -> LDS(AS3) addrspacecast; AS3 pointers are the 32-bit LDS offset
  return (unsigned)(unsigned long long)
         (__attribute__((address_space(3))) const char*)(const char*)p;
}

__device__ __forceinline__ void async_copy_b128(unsigned lds_off, const void* gsrc) {
  asm volatile("global_load_async_to_lds_b128 %0, %1, off"
               :: "v"(lds_off), "v"((unsigned long long)gsrc)
               : "memory");
}

// -------------------------------------------------------------------------
// f32 -> bf16 conversion
// -------------------------------------------------------------------------
__global__ __launch_bounds__(256)
void f32_to_bf16_kernel(const float* __restrict__ in, __bf16* __restrict__ out, int n) {
  int i = blockIdx.x * blockDim.x + threadIdx.x;
  if (i < n) out[i] = (__bf16)in[i];
}

// -------------------------------------------------------------------------
// Projection GEMM: out = scale * (A @ W^T).
// A:[M][E] bf16 row-major, W:[E][E] bf16 row-major (NxK: B-columns are W rows).
// Each wave computes a 32x64 tile (2 A-frags reused over 4 B-frags -> 8 WMMA
// per k-step); 4 waves/block span 128 rows.
// VT=false: out bf16 row-major [M][E];  VT=true: V path -> Vt[b][h][d][tc]
// -------------------------------------------------------------------------
template<bool VT>
__global__ __launch_bounds__(128)
void proj_gemm_kernel(const __bf16* __restrict__ A, const __bf16* __restrict__ W,
                      __bf16* __restrict__ out, float scale) {
  const int lane = threadIdx.x & 31;
  const int wave = threadIdx.x >> 5;
  const int lo = lane & 15, hi = lane >> 4;
  const int m0 = blockIdx.x * 128 + wave * 32;
  const int n0 = blockIdx.y * 64;

  v8f acc[2][4] = {};
  for (int k0 = 0; k0 < E_DIM; k0 += 32) {
    v16bf af[2];
#pragma unroll
    for (int mb = 0; mb < 2; ++mb) {
      const __bf16* arow = A + (size_t)(m0 + mb * 16 + lo) * E_DIM + k0 + hi * 8;
#pragma unroll
      for (int i = 0; i < 8; ++i) { af[mb][i] = arow[i]; af[mb][8 + i] = arow[16 + i]; }
    }
#pragma unroll
    for (int nb = 0; nb < 4; ++nb) {
      const __bf16* wrow = W + (size_t)(n0 + nb * 16 + lo) * E_DIM + k0 + hi * 8;
      v16bf bfr;
#pragma unroll
      for (int i = 0; i < 8; ++i) { bfr[i] = wrow[i]; bfr[8 + i] = wrow[16 + i]; }
#pragma unroll
      for (int mb = 0; mb < 2; ++mb) {
        acc[mb][nb] = __builtin_amdgcn_wmma_f32_16x16x32_bf16(
            false, af[mb], false, bfr, (short)0, acc[mb][nb], false, false);
      }
    }
  }

#pragma unroll
  for (int mb = 0; mb < 2; ++mb) {
#pragma unroll
    for (int nb = 0; nb < 4; ++nb) {
#pragma unroll
      for (int r = 0; r < 8; ++r) {
        int m = m0 + mb * 16 + r + 8 * hi;
        int n = n0 + nb * 16 + lo;
        __bf16 v = (__bf16)(acc[mb][nb][r] * scale);
        if constexpr (!VT) {
          out[(size_t)m * E_DIM + n] = v;
        } else {
          int bi = m >> 11;                 // / TC_DIM
          int c  = m & (TC_DIM - 1);
          int h  = n >> 6;                  // / HDIM
          int d  = n & (HDIM - 1);
          out[(size_t)((bi * HEADS + h) * HDIM + d) * TC_DIM + c] = v;
        }
      }
    }
  }
}

// -------------------------------------------------------------------------
// Flash attention. Block = 8 waves sharing one (b,h); each wave owns 16
// query rows. K/V chunks (32 keys) staged in LDS once per block via
// double-buffered async global->LDS copies. Softmax denominator kept as
// per-lane partials (rescale factor is lane-uniform), reduced once at end.
// -------------------------------------------------------------------------
__global__ __launch_bounds__(256)
void flash_attn_kernel(const __bf16* __restrict__ Q, const __bf16* __restrict__ Kp,
                       const __bf16* __restrict__ Vt, __bf16* __restrict__ AO) {
  __shared__ __align__(16) __bf16 kbuf[2][32 * 64];  // [key row][d]      4KB each
  __shared__ __align__(16) __bf16 vbuf[2][64 * 32];  // [d][key row]      4KB each
  __shared__ __align__(16) __bf16 pbuf[8][16 * 32];  // per-wave P stage  1KB each

  const int tid  = threadIdx.x;
  const int lane = tid & 31;
  const int wave = tid >> 5;
  const int lo = lane & 15, hi = lane >> 4;
  const int bh = blockIdx.x;                 // b*HEADS + h
  const int bi = bh / HEADS, h = bh % HEADS;
  const int q0 = blockIdx.y * 128 + wave * 16;

  const __bf16* kBase = Kp + (size_t)bi * TC_DIM * E_DIM + h * HDIM;       // + c*E + d
  const __bf16* vBase = Vt + (size_t)(bi * HEADS + h) * HDIM * TC_DIM;     // + d*TC + c

  // async staging of K/V chunk starting at key c0n into buffer `buf`
  // (256 threads: one 16B K-segment + one 16B V-segment each -> 2 async/wave)
  auto stage_chunk = [&](int c0n, int buf) {
    unsigned kdst = lds_addr(&kbuf[buf][0]);
    unsigned vdst = lds_addr(&vbuf[buf][0]);
    int kr = tid >> 3, kc = tid & 7;         // K: 32 rows x 8 segs
    async_copy_b128(kdst + (unsigned)(kr * 64 + kc * 8) * 2u,
                    kBase + (size_t)(c0n + kr) * E_DIM + kc * 8);
    int vr = tid >> 2, vc = tid & 3;         // V: 64 rows x 4 segs
    async_copy_b128(vdst + (unsigned)(vr * 32 + vc * 8) * 2u,
                    vBase + (size_t)vr * TC_DIM + c0n + vc * 8);
  };

  // Q fragments for d-blocks {0..31},{32..63} (A-matrix layout)
  v16bf qf[2];
#pragma unroll
  for (int db = 0; db < 2; ++db) {
    const __bf16* qrow = Q + (size_t)(bi * T_DIM + q0 + lo) * E_DIM + h * HDIM + db * 32 + hi * 8;
#pragma unroll
    for (int i = 0; i < 8; ++i) { qf[db][i] = qrow[i]; qf[db][8 + i] = qrow[16 + i]; }
  }

  v8f acc[4] = {};
  float mrun[8], lpart[8];                   // running max + per-lane partial sums
#pragma unroll
  for (int r = 0; r < 8; ++r) { mrun[r] = -3.0e38f; lpart[r] = 0.0f; }

  __bf16* myp = pbuf[wave];

  stage_chunk(0, 0);                         // preload first chunk (2 async ops/wave)

  for (int c0 = 0; c0 < TC_DIM; c0 += 32) {
    const int buf = (c0 >> 5) & 1;
    if (c0 + 32 < TC_DIM) {
      stage_chunk(c0 + 32, buf ^ 1);         // stream next chunk into other buffer
      asm volatile("s_wait_asynccnt 0x2" ::: "memory");   // current chunk landed
    } else {
      asm volatile("s_wait_asynccnt 0x0" ::: "memory");
    }
    __syncthreads();                         // all waves' staging visible

    const __bf16* kb = &kbuf[buf][0];
    const __bf16* vb = &vbuf[buf][0];

    // ---- scores: S[16][32] = Q(16x64) . K^T (fragments from LDS) ----
    v8f s[2] = {};
#pragma unroll
    for (int ns = 0; ns < 2; ++ns) {
#pragma unroll
      for (int db = 0; db < 2; ++db) {
        const __bf16* krow = kb + (ns * 16 + lo) * 64 + db * 32 + hi * 8;
        v16bf kf;
#pragma unroll
        for (int i = 0; i < 8; ++i) { kf[i] = krow[i]; kf[8 + i] = krow[16 + i]; }
        s[ns] = __builtin_amdgcn_wmma_f32_16x16x32_bf16(false, qf[db], false, kf,
                                                        (short)0, s[ns], false, false);
      }
    }

    // ---- online softmax: only the row-MAX needs a cross-lane reduction;
    //      the denominator stays a per-lane partial (sc is lane-uniform) ----
#pragma unroll
    for (int r = 0; r < 8; ++r) {
      float cur = fmaxf(s[0][r], s[1][r]);
#pragma unroll
      for (int msk = 1; msk < 16; msk <<= 1) cur = fmaxf(cur, __shfl_xor(cur, msk, 32));
      float nm = fmaxf(mrun[r], cur);
      float sc = __expf(mrun[r] - nm);
      float p0 = __expf(s[0][r] - nm);
      float p1 = __expf(s[1][r] - nm);
      lpart[r] = lpart[r] * sc + p0 + p1;    // this lane's 2 columns only
      mrun[r] = nm;
#pragma unroll
      for (int ds = 0; ds < 4; ++ds) acc[ds][r] *= sc;
      myp[(r + 8 * hi) * 32 + lo]      = (__bf16)p0;
      myp[(r + 8 * hi) * 32 + lo + 16] = (__bf16)p1;
    }
    // pbuf is wave-private: DS-counter ordering suffices (no block barrier)
    asm volatile("s_wait_dscnt 0x0" ::: "memory");

    // reload P in A-matrix fragment layout
    v16bf pf;
    const __bf16* prow = myp + lo * 32 + hi * 8;
#pragma unroll
    for (int i = 0; i < 8; ++i) { pf[i] = prow[i]; pf[8 + i] = prow[16 + i]; }

    // ---- accumulate O += P(16x32) . V(32x64) (V fragments from LDS) ----
#pragma unroll
    for (int ds = 0; ds < 4; ++ds) {
      const __bf16* vrow = vb + (ds * 16 + lo) * 32 + hi * 8;
      v16bf vf;
#pragma unroll
      for (int i = 0; i < 8; ++i) { vf[i] = vrow[i]; vf[8 + i] = vrow[16 + i]; }
      acc[ds] = __builtin_amdgcn_wmma_f32_16x16x32_bf16(false, pf, false, vf,
                                                        (short)0, acc[ds], false, false);
    }
    __syncthreads();                         // buffers free before next overwrite
  }

  // ---- single deferred denominator reduction, then normalize + store ----
#pragma unroll
  for (int r = 0; r < 8; ++r) {
    float l = lpart[r];
#pragma unroll
    for (int msk = 1; msk < 16; msk <<= 1) l += __shfl_xor(l, msk, 32);
    float inv = 1.0f / l;
    int trow = bi * T_DIM + q0 + r + 8 * hi;
#pragma unroll
    for (int ds = 0; ds < 4; ++ds) {
      AO[(size_t)trow * E_DIM + h * HDIM + ds * 16 + lo] = (__bf16)(acc[ds][r] * inv);
    }
  }
}

// -------------------------------------------------------------------------
// Output projection: Y = AO @ Wo^T + bo, f32 result into d_out.
// -------------------------------------------------------------------------
__global__ __launch_bounds__(128)
void out_proj_kernel(const __bf16* __restrict__ AO, const __bf16* __restrict__ Wo,
                     const float* __restrict__ bo, float* __restrict__ Y) {
  const int lane = threadIdx.x & 31;
  const int wave = threadIdx.x >> 5;
  const int lo = lane & 15, hi = lane >> 4;
  const int m0 = blockIdx.x * 128 + wave * 32;
  const int n0 = blockIdx.y * 64;

  v8f acc[2][4] = {};
  for (int k0 = 0; k0 < E_DIM; k0 += 32) {
    v16bf af[2];
#pragma unroll
    for (int mb = 0; mb < 2; ++mb) {
      const __bf16* arow = AO + (size_t)(m0 + mb * 16 + lo) * E_DIM + k0 + hi * 8;
#pragma unroll
      for (int i = 0; i < 8; ++i) { af[mb][i] = arow[i]; af[mb][8 + i] = arow[16 + i]; }
    }
#pragma unroll
    for (int nb = 0; nb < 4; ++nb) {
      const __bf16* wrow = Wo + (size_t)(n0 + nb * 16 + lo) * E_DIM + k0 + hi * 8;
      v16bf bfr;
#pragma unroll
      for (int i = 0; i < 8; ++i) { bfr[i] = wrow[i]; bfr[8 + i] = wrow[16 + i]; }
#pragma unroll
      for (int mb = 0; mb < 2; ++mb) {
        acc[mb][nb] = __builtin_amdgcn_wmma_f32_16x16x32_bf16(
            false, af[mb], false, bfr, (short)0, acc[mb][nb], false, false);
      }
    }
  }

#pragma unroll
  for (int nb = 0; nb < 4; ++nb) {
    float bias = bo[n0 + nb * 16 + lo];
#pragma unroll
    for (int mb = 0; mb < 2; ++mb) {
#pragma unroll
      for (int r = 0; r < 8; ++r) {
        Y[(size_t)(m0 + mb * 16 + r + 8 * hi) * E_DIM + n0 + nb * 16 + lo] =
            acc[mb][nb][r] + bias;
      }
    }
  }
}

// -------------------------------------------------------------------------
// Launch
// -------------------------------------------------------------------------
extern "C" void kernel_launch(void* const* d_in, const int* in_sizes, int n_in,
                              void* d_out, int out_size, void* d_ws, size_t ws_size,
                              hipStream_t stream) {
  const float* x   = (const float*)d_in[0];
  const float* ctx = (const float*)d_in[1];
  const float* Wq  = (const float*)d_in[2];
  const float* Wk  = (const float*)d_in[3];
  const float* Wv  = (const float*)d_in[4];
  const float* Wo  = (const float*)d_in[5];
  const float* bo  = (const float*)d_in[6];
  float* out = (float*)d_out;

  char* ws = (char*)d_ws;
  size_t off = 0;
  auto alloc = [&](size_t bytes) -> void* {
    void* p = ws + off;
    off += (bytes + 255) & ~(size_t)255;
    return p;
  };

  const size_t MN = (size_t)BATCH * T_DIM * E_DIM;   // 4M elems
  const size_t WN = (size_t)E_DIM * E_DIM;           // 1M elems
  __bf16* xb  = (__bf16*)alloc(MN * 2);
  __bf16* cb  = (__bf16*)alloc(MN * 2);
  __bf16* wqb = (__bf16*)alloc(WN * 2);
  __bf16* wkb = (__bf16*)alloc(WN * 2);
  __bf16* wvb = (__bf16*)alloc(WN * 2);
  __bf16* wob = (__bf16*)alloc(WN * 2);
  __bf16* Qb  = (__bf16*)alloc(MN * 2);
  __bf16* Kb  = (__bf16*)alloc(MN * 2);
  __bf16* Vtb = (__bf16*)alloc(MN * 2);
  __bf16* AOb = (__bf16*)alloc(MN * 2);

  // 1) conversions
  f32_to_bf16_kernel<<<(int)((MN + 255) / 256), 256, 0, stream>>>(x, xb, (int)MN);
  f32_to_bf16_kernel<<<(int)((MN + 255) / 256), 256, 0, stream>>>(ctx, cb, (int)MN);
  f32_to_bf16_kernel<<<(int)((WN + 255) / 256), 256, 0, stream>>>(Wq, wqb, (int)WN);
  f32_to_bf16_kernel<<<(int)((WN + 255) / 256), 256, 0, stream>>>(Wk, wkb, (int)WN);
  f32_to_bf16_kernel<<<(int)((WN + 255) / 256), 256, 0, stream>>>(Wv, wvb, (int)WN);
  f32_to_bf16_kernel<<<(int)((WN + 255) / 256), 256, 0, stream>>>(Wo, wob, (int)WN);

  // 2) projections (Q,K pre-scaled by E^-0.25 = 1024^-0.25)
  const float inv4 = 0.17677669529663687f;
  dim3 gproj(BATCH * T_DIM / 128, E_DIM / 64);       // (32, 16)
  proj_gemm_kernel<false><<<gproj, 128, 0, stream>>>(xb, wqb, Qb,  inv4);
  proj_gemm_kernel<false><<<gproj, 128, 0, stream>>>(cb, wkb, Kb,  inv4);
  proj_gemm_kernel<true ><<<gproj, 128, 0, stream>>>(cb, wvb, Vtb, 1.0f);

  // 3) fused flash attention with async LDS staging (8 waves/block)
  dim3 gattn(BATCH * HEADS, T_DIM / 128);            // (32, 16)
  flash_attn_kernel<<<gattn, 256, 0, stream>>>(Qb, Kb, Vtb, AOb);

  // 4) output projection + bias -> f32 d_out
  out_proj_kernel<<<gproj, 128, 0, stream>>>(AOb, wob, bo, out);
}
